// PETformerBlock_14259291423211
// MI455X (gfx1250) — matrix operand order repaired and checked by reading, not compiled, not run on hardware
//
#include <hip/hip_runtime.h>
#include <hip/hip_bf16.h>
#include <math.h>

// ---------------------------------------------------------------------------
// PETformer block for MI455X (gfx1250, wave32).
// Heavy op: per-pixel dynamic-kernel GEMM (a @ kw, K=nset=32) done with
// v_wmma_f32_16x16x32_f16, fused with the KBA contraction + GELU gate.
// All intermediates (~70MB fp32) stay L2-resident (192MB).
// ---------------------------------------------------------------------------

typedef _Float16 f16;
typedef __attribute__((ext_vector_type(16))) _Float16 v16h;
typedef __attribute__((ext_vector_type(8)))  float    v8f;

#define Bn    4
#define Cc    48
#define HWp   4096      // 64*64
#define HIDc  96
#define NSETc 32
#define INTERCc 24
#define Np    16384     // Bn*HWp

// ---------------------------------------------------------------------------
// LayerNorm2d over channels, producing both branch inputs xn (n1) and xm (n2)
// ---------------------------------------------------------------------------
__global__ __launch_bounds__(256) void ln_kernel(
    const float* __restrict__ x,
    const float* __restrict__ n1w, const float* __restrict__ n1b,
    const float* __restrict__ n2w, const float* __restrict__ n2b,
    float* __restrict__ xn, float* __restrict__ xm)
{
    int P = blockIdx.x * 256 + threadIdx.x;     // global pixel
    int b = P >> 12, p = P & 4095;
    float v[Cc];
    float mu = 0.f;
#pragma unroll
    for (int c = 0; c < Cc; ++c) { v[c] = x[((size_t)(b*Cc + c) << 12) + p]; mu += v[c]; }
    mu *= (1.f / Cc);
    float var = 0.f;
#pragma unroll
    for (int c = 0; c < Cc; ++c) { float d = v[c] - mu; var += d * d; }
    var *= (1.f / Cc);
    float inv = rsqrtf(var + 1e-6f);
#pragma unroll
    for (int c = 0; c < Cc; ++c) {
        float y = (v[c] - mu) * inv;
        size_t o = ((size_t)(b*Cc + c) << 12) + p;
        xn[o] = n1w[c] * y + n1b[c];
        xm[o] = n2w[c] * y + n2b[c];
    }
}

// ---------------------------------------------------------------------------
// Global average pool per (b,c)
// ---------------------------------------------------------------------------
__global__ __launch_bounds__(256) void gap_kernel(const float* __restrict__ x,
                                                  float* __restrict__ gap)
{
    __shared__ float red[256];
    int bc = blockIdx.x;                         // b*48 + c
    float s = 0.f;
    for (int p = threadIdx.x; p < HWp; p += 256) s += x[((size_t)bc << 12) + p];
    red[threadIdx.x] = s; __syncthreads();
    for (int st = 128; st > 0; st >>= 1) {
        if (threadIdx.x < st) red[threadIdx.x] += red[threadIdx.x + st];
        __syncthreads();
    }
    if (threadIdx.x == 0) gap[bc] = red[0] * (1.f / HWp);
}

// Channel attention 1x1 convs on pooled vector: ca1, ca2 (B*C outputs each)
__global__ void ca_kernel(const float* __restrict__ gap,
                          const float* __restrict__ w1, const float* __restrict__ b1,
                          const float* __restrict__ w2, const float* __restrict__ b2,
                          float* __restrict__ ca1, float* __restrict__ ca2)
{
    int t = threadIdx.x;
    if (t < Bn * Cc) {
        int b = t / Cc, co = t % Cc;
        float a1 = b1[co], a2 = b2[co];
        for (int ci = 0; ci < Cc; ++ci) {
            float g = gap[b*Cc + ci];
            a1 = fmaf(w1[co*Cc + ci], g, a1);
            a2 = fmaf(w2[co*Cc + ci], g, a2);
        }
        ca1[t] = a1; ca2[t] = a2;
    }
}

// ---------------------------------------------------------------------------
// Generic 1x1 conv (NCHW): out[b,cob+j,p] = bias + sum_ci W[co,ci]*in[b,ci,p]
// blockIdx.y selects a chunk of 16 output channels; weights staged in LDS.
// ---------------------------------------------------------------------------
__global__ __launch_bounds__(256) void conv1x1_kernel(
    const float* __restrict__ in, const float* __restrict__ W,
    const float* __restrict__ bias, float* __restrict__ out,
    int Cin, int Cout)
{
    __shared__ float sw[16 * 96];
    const int cob = blockIdx.y * 16;
    for (int t = threadIdx.x; t < 16 * Cin; t += 256) sw[t] = W[cob * Cin + t];
    __syncthreads();
    int P = blockIdx.x * 256 + threadIdx.x;
    int b = P >> 12, p = P & 4095;
    float acc[16];
#pragma unroll
    for (int j = 0; j < 16; ++j) acc[j] = bias ? bias[cob + j] : 0.f;
    for (int ci = 0; ci < Cin; ++ci) {
        float v = in[((size_t)(b*Cin + ci) << 12) + p];
#pragma unroll
        for (int j = 0; j < 16; ++j) acc[j] = fmaf(sw[j*Cin + ci], v, acc[j]);
    }
#pragma unroll
    for (int j = 0; j < 16; ++j)
        out[((size_t)(b*Cout + cob + j) << 12) + p] = acc[j];
}

// ---------------------------------------------------------------------------
// Depthwise 3x3 conv, pad=1, no bias.  idx over b*Ch*HW.
// ---------------------------------------------------------------------------
__global__ __launch_bounds__(256) void dw3x3_kernel(
    const float* __restrict__ in, const float* __restrict__ w,
    float* __restrict__ out, int Ch)
{
    int idx = blockIdx.x * 256 + threadIdx.x;
    int p = idx & 4095; int bc = idx >> 12;
    int c = bc % Ch;
    int h = p >> 6, wc = p & 63;
    float acc = 0.f;
#pragma unroll
    for (int di = 0; di < 3; ++di)
#pragma unroll
        for (int dj = 0; dj < 3; ++dj) {
            int r = h + di - 1, ccol = wc + dj - 1;
            if (r >= 0 && r < 64 && ccol >= 0 && ccol < 64)
                acc = fmaf(in[((size_t)bc << 12) + r*64 + ccol], w[c*9 + di*3 + dj], acc);
        }
    out[idx] = acc;
}

// ---------------------------------------------------------------------------
// Grouped 3x3 conv for c2a: 48->24 ch, groups=24 (2 in per group), pad=1, bias
// ---------------------------------------------------------------------------
__global__ __launch_bounds__(256) void gconv3x3_kernel(
    const float* __restrict__ in, const float* __restrict__ w,
    const float* __restrict__ bias, float* __restrict__ out)
{
    int idx = blockIdx.x * 256 + threadIdx.x;   // over B*24*HW
    int p = idx & 4095; int bgo = idx >> 12;
    int go = bgo % INTERCc; int b = bgo / INTERCc;
    int h = p >> 6, wc = p & 63;
    float acc = bias[go];
#pragma unroll
    for (int ci = 0; ci < 2; ++ci)
#pragma unroll
        for (int di = 0; di < 3; ++di)
#pragma unroll
            for (int dj = 0; dj < 3; ++dj) {
                int r = h + di - 1, ccol = wc + dj - 1;
                if (r >= 0 && r < 64 && ccol >= 0 && ccol < 64)
                    acc = fmaf(in[((size_t)(b*Cc + go*2 + ci) << 12) + r*64 + ccol],
                               w[(go*2 + ci)*9 + di*3 + dj], acc);
            }
    out[idx] = acc;
}

// ---------------------------------------------------------------------------
// SimpleGate + c2b 1x1 + attgamma + c211 1x1  ->  att, stored as f16 in the
// [b][nset][p] layout the WMMA kernel consumes directly.
// ---------------------------------------------------------------------------
__global__ __launch_bounds__(256) void att_kernel(
    const float* __restrict__ apre, const float* __restrict__ xn,
    const float* __restrict__ c2bw, const float* __restrict__ c2bb,
    const float* __restrict__ c211w, const float* __restrict__ c211b,
    const float* __restrict__ attg, f16* __restrict__ atth)
{
    __shared__ float s2b[NSETc * 12], s211[NSETc * Cc];
    __shared__ float sb2[NSETc], sb1[NSETc], sgam[NSETc];
    for (int t = threadIdx.x; t < NSETc * 12; t += 256) s2b[t]  = c2bw[t];
    for (int t = threadIdx.x; t < NSETc * Cc; t += 256) s211[t] = c211w[t];
    if (threadIdx.x < NSETc) {
        sb2[threadIdx.x] = c2bb[threadIdx.x];
        sb1[threadIdx.x] = c211b[threadIdx.x];
        sgam[threadIdx.x] = attg[threadIdx.x];
    }
    __syncthreads();
    int P = blockIdx.x * 256 + threadIdx.x;
    int b = P >> 12, p = P & 4095;
    float sgv[12], xv[Cc];
#pragma unroll
    for (int ci = 0; ci < 12; ++ci)
        sgv[ci] = apre[((size_t)(b*INTERCc + ci) << 12) + p] *
                  apre[((size_t)(b*INTERCc + 12 + ci) << 12) + p];
#pragma unroll
    for (int c = 0; c < Cc; ++c) xv[c] = xn[((size_t)(b*Cc + c) << 12) + p];
    for (int n = 0; n < NSETc; ++n) {
        float a = sb2[n];
#pragma unroll
        for (int ci = 0; ci < 12; ++ci) a = fmaf(s2b[n*12 + ci], sgv[ci], a);
        float o = sb1[n];
#pragma unroll
        for (int c = 0; c < Cc; ++c) o = fmaf(s211[n*Cc + c], xv[c], o);
        atth[((size_t)(b*NSETc + n) << 12) + p] = (f16)(a * sgam[n] + o);
    }
}

// fp32 -> f16 weight conversion (kw / kb staged once, L2-resident)
__global__ void cvt_kernel(const float* __restrict__ in, f16* __restrict__ out, int n)
{
    int i = blockIdx.x * 256 + threadIdx.x;
    if (i < n) out[i] = (f16)in[i];
}

// ---------------------------------------------------------------------------
// KBA core: for each 16-pixel tile (one per wave), generate the per-pixel
// dynamic kernels with v_wmma_f32_16x16x32_f16 (K = nset = 32 exactly),
// stage to LDS, contract with the 3x3 im2col patch, fuse *ga1 + uf and the
// GELU(x1) gate.  Output x2g is the input of the kproj 1x1 conv.
// ---------------------------------------------------------------------------
#define KBA_WAVES 4
__global__ __launch_bounds__(128) void kba_kernel(
    const f16* __restrict__ atth, const f16* __restrict__ kwh,
    const f16* __restrict__ kbh,  const float* __restrict__ ufb,
    const float* __restrict__ x1, const float* __restrict__ ga1,
    float* __restrict__ x2g)
{
    __shared__ float s_bias[KBA_WAVES][16][96];   // 24 KB
    __shared__ float s_k[KBA_WAVES][16][144];     // 36 KB
    __shared__ float s_patch[KBA_WAVES][224];     // ~3.5 KB

    const int lane = threadIdx.x & 31;
    const int wv   = threadIdx.x >> 5;
    const int tile = blockIdx.x * KBA_WAVES + wv;  // 0..1023
    const int b    = tile >> 8;                    // 256 tiles per image
    const int pimg = (tile & 255) * 16;            // base pixel in image
    const int hrow = pimg >> 6;
    const int w0   = pimg & 63;                    // multiple of 16 -> one row
    const int m    = lane & 15;
    const int hseg = lane >> 4;

    // A tile (16 px x 32 nset), contiguous K order matching B:
    // lanes 0-15 hold K {0..15}, lanes 16-31 hold K {16..31}
    v16h A;
    const int p = pimg + m;
#pragma unroll
    for (int e = 0; e < 16; ++e) {
        int k = hseg * 16 + e;
        A[e] = atth[((size_t)(b * NSETc + k) << 12) + p];
    }

    // bias = a @ kb  (32x96): 6 WMMA column tiles -> LDS
#pragma unroll
    for (int cb = 0; cb < 6; ++cb) {
        v16h Bm;
#pragma unroll
        for (int e = 0; e < 16; ++e) {
            int k = hseg * 16 + e;                 // contiguous K order, same as A
            Bm[e] = kbh[k * HIDc + cb * 16 + m];
        }
        v8f D = {};
        D = __builtin_amdgcn_wmma_f32_16x16x32_f16(false, A, false, Bm,
                                                   (short)0, D, false, false);
#pragma unroll
        for (int r = 0; r < 8; ++r)
            s_bias[wv][r + hseg * 8][cb * 16 + m] = D[r];
    }
    __syncthreads();

    for (int g = 0; g < 24; ++g) {
        // dynamic kernels for this group: 9 WMMAs over 144 columns of kw
#pragma unroll
        for (int cc = 0; cc < 9; ++cc) {
            v16h Bm;
#pragma unroll
            for (int e = 0; e < 16; ++e) {
                int k = hseg * 16 + e;
                Bm[e] = kwh[k * 3456 + g * 144 + cc * 16 + m];
            }
            v8f D = {};
            D = __builtin_amdgcn_wmma_f32_16x16x32_f16(false, A, false, Bm,
                                                       (short)0, D, false, false);
#pragma unroll
            for (int r = 0; r < 8; ++r)
                s_k[wv][r + hseg * 8][cc * 16 + m] = D[r];
        }
        // im2col patch for this group: 4 ch x 3 rows x 18 cols (zero-padded)
        for (int t = lane; t < 216; t += 32) {
            int ci = t / 54, rem = t % 54;
            int di = rem / 18, dc = rem % 18;
            int rr = hrow + di - 1, ccol = w0 + dc - 1;
            float v = 0.f;
            if (rr >= 0 && rr < 64 && ccol >= 0 && ccol < 64)
                v = ufb[((size_t)(b * HIDc + g * 4 + ci) << 12) + rr * 64 + ccol];
            s_patch[wv][t] = v;
        }
        __syncthreads();

        // contraction: each lane produces 2 of the 64 (pixel, i) outputs
#pragma unroll
        for (int s = 0; s < 2; ++s) {
            int q  = lane * 2 + s;
            int px = q >> 2, i = q & 3;
            int c  = g * 4 + i;
            float acc = s_bias[wv][px][c];
#pragma unroll
            for (int j = 0; j < 36; ++j) {
                int ci = j / 9, kk = j % 9;
                int di = kk / 3, dj = kk % 3;
                acc = fmaf(s_k[wv][px][i * 36 + j],
                           s_patch[wv][ci * 54 + di * 18 + px + dj], acc);
            }
            float ufc = s_patch[wv][i * 54 + 18 + px + 1];   // center tap
            float x2  = acc * ga1[c] + ufc;
            size_t oidx = ((size_t)(b * HIDc + c) << 12) + pimg + px;
            float xv  = x1[oidx];
            float gel = 0.5f * xv * (1.0f + erff(xv * 0.70710678118f)); // exact GELU
            x2g[oidx] = gel * x2;
        }
        __syncthreads();
    }
}

// ---------------------------------------------------------------------------
// MDTA (transposed channel attention): one block per (b, head).
// ---------------------------------------------------------------------------
__global__ __launch_bounds__(256) void mdta_kernel(
    const float* __restrict__ qkv, const float* __restrict__ temp,
    float* __restrict__ out)
{
    __shared__ float part[256][36];
    __shared__ float inv[12];
    __shared__ float attn[6][6];
    int bh = blockIdx.x;
    int b = bh >> 3, h = bh & 7;
    const float* qb = qkv + (((size_t)b * 144 + h * 6) << 12);
    const float* kb_ = qkv + (((size_t)b * 144 + 48 + h * 6) << 12);
    const float* vb = qkv + (((size_t)b * 144 + 96 + h * 6) << 12);
    int tid = threadIdx.x;

    // q/k row norms (12 rows of 4096)
    for (int r = 0; r < 12; ++r) {
        const float* src = (r < 6) ? (qb + ((size_t)r << 12)) : (kb_ + ((size_t)(r - 6) << 12));
        float s = 0.f;
        for (int p = tid; p < HWp; p += 256) { float v = src[p]; s += v * v; }
        part[tid][0] = s; __syncthreads();
        if (tid < 128) part[tid][0] += part[tid + 128][0]; __syncthreads();
        if (tid <  64) part[tid][0] += part[tid +  64][0]; __syncthreads();
        if (tid <  32) part[tid][0] += part[tid +  32][0]; __syncthreads();
        if (tid == 0) {
            float tot = 0.f;
            for (int i = 0; i < 32; ++i) tot += part[i][0];
            inv[r] = 1.f / fmaxf(sqrtf(tot), 1e-12f);
        }
        __syncthreads();
    }

    // S = q k^T (6x6), each thread accumulates a p-slice
    float acc[36];
#pragma unroll
    for (int i = 0; i < 36; ++i) acc[i] = 0.f;
    for (int p = tid; p < HWp; p += 256) {
        float qv[6], kv[6];
#pragma unroll
        for (int d = 0; d < 6; ++d) {
            qv[d] = qb[((size_t)d << 12) + p];
            kv[d] = kb_[((size_t)d << 12) + p];
        }
#pragma unroll
        for (int c = 0; c < 6; ++c)
#pragma unroll
            for (int d = 0; d < 6; ++d)
                acc[c * 6 + d] = fmaf(qv[c], kv[d], acc[c * 6 + d]);
    }
#pragma unroll
    for (int i = 0; i < 36; ++i) part[tid][i] = acc[i];
    __syncthreads();
    if (tid < 36) {
        float s = 0.f;
        for (int t = 0; t < 256; ++t) s += part[t][tid];
        int c = tid / 6, d = tid % 6;
        part[0][tid] = s * inv[c] * inv[6 + d] * temp[h];
    }
    __syncthreads();
    if (tid < 6) {
        int c = tid;
        float mx = -1e30f;
        for (int d = 0; d < 6; ++d) mx = fmaxf(mx, part[0][c * 6 + d]);
        float e[6], sum = 0.f;
        for (int d = 0; d < 6; ++d) { e[d] = expf(part[0][c * 6 + d] - mx); sum += e[d]; }
        for (int d = 0; d < 6; ++d) attn[c][d] = e[d] / sum;
    }
    __syncthreads();
    float aw[36];
#pragma unroll
    for (int i = 0; i < 36; ++i) aw[i] = attn[i / 6][i % 6];
    float* ob = out + (((size_t)b * Cc + h * 6) << 12);
    for (int p = tid; p < HWp; p += 256) {
        float vv[6];
#pragma unroll
        for (int d = 0; d < 6; ++d) vv[d] = vb[((size_t)d << 12) + p];
#pragma unroll
        for (int c = 0; c < 6; ++c) {
            float o = 0.f;
#pragma unroll
            for (int d = 0; d < 6; ++d) o = fmaf(aw[c * 6 + d], vv[d], o);
            ob[((size_t)c << 12) + p] = o;
        }
    }
}

// Final residual combine: out = x + kba*ca1[b,c] + mdta*ca2[b,c]
__global__ __launch_bounds__(256) void combine_kernel(
    const float* __restrict__ x, const float* __restrict__ kba,
    const float* __restrict__ mdta, const float* __restrict__ ca1,
    const float* __restrict__ ca2, float* __restrict__ out)
{
    int idx = blockIdx.x * 256 + threadIdx.x;
    int bc = idx >> 12;
    out[idx] = x[idx] + kba[idx] * ca1[bc] + mdta[idx] * ca2[bc];
}

// ---------------------------------------------------------------------------
extern "C" void kernel_launch(void* const* d_in, const int* in_sizes, int n_in,
                              void* d_out, int out_size, void* d_ws, size_t ws_size,
                              hipStream_t stream)
{
    (void)in_sizes; (void)n_in; (void)out_size; (void)ws_size;
    const float* x      = (const float*)d_in[0];
    const float* n1w    = (const float*)d_in[1];
    const float* n1b    = (const float*)d_in[2];
    const float* n2w    = (const float*)d_in[3];
    const float* n2b    = (const float*)d_in[4];
    const float* kdw1   = (const float*)d_in[5];
    const float* kdw2   = (const float*)d_in[6];
    const float* kc1a   = (const float*)d_in[7];
    const float* kc1b   = (const float*)d_in[8];
    const float* kproj  = (const float*)d_in[9];
    const float* c2aw   = (const float*)d_in[10];
    const float* c2ab   = (const float*)d_in[11];
    const float* c2bw   = (const float*)d_in[12];
    const float* c2bb   = (const float*)d_in[13];
    const float* c211w  = (const float*)d_in[14];
    const float* c211b  = (const float*)d_in[15];
    const float* kw     = (const float*)d_in[16];
    const float* kb     = (const float*)d_in[17];
    const float* attg   = (const float*)d_in[18];
    const float* ga1    = (const float*)d_in[19];
    const float* temp   = (const float*)d_in[20];
    const float* qkvw   = (const float*)d_in[21];
    const float* qkvdw  = (const float*)d_in[22];
    const float* mprojw = (const float*)d_in[23];
    const float* ca1w   = (const float*)d_in[24];
    const float* ca1b   = (const float*)d_in[25];
    const float* ca2w   = (const float*)d_in[26];
    const float* ca2b   = (const float*)d_in[27];

    // ---- workspace layout (bytes, 256-aligned); total ~70 MB (L2-resident)
    char* ws = (char*)d_ws;
    size_t off = 0;
    auto alloc = [&](size_t bytes) -> void* {
        void* q = ws + off;
        off = (off + bytes + 255) & ~(size_t)255;
        return q;
    };
    float* xn     = (float*)alloc((size_t)Np * Cc   * 4);
    float* xm     = (float*)alloc((size_t)Np * Cc   * 4);
    float* x1p    = (float*)alloc((size_t)Np * HIDc * 4);
    float* x1     = (float*)alloc((size_t)Np * HIDc * 4);
    float* ufp    = (float*)alloc((size_t)Np * HIDc * 4);
    float* ufb    = (float*)alloc((size_t)Np * HIDc * 4);
    float* apre   = (float*)alloc((size_t)Np * INTERCc * 4);
    f16*   atth   = (f16*)  alloc((size_t)Np * NSETc * 2);
    f16*   kwh    = (f16*)  alloc((size_t)NSETc * 3456 * 2);
    f16*   kbh    = (f16*)  alloc((size_t)NSETc * HIDc * 2);
    float* x2g    = (float*)alloc((size_t)Np * HIDc * 4);
    float* kbaout = (float*)alloc((size_t)Np * Cc   * 4);
    float* qkvp   = (float*)alloc((size_t)Np * 144  * 4);
    float* qkv    = (float*)alloc((size_t)Np * 144  * 4);
    float* attno  = (float*)alloc((size_t)Np * Cc   * 4);
    float* mdtao  = (float*)alloc((size_t)Np * Cc   * 4);
    float* gapb   = (float*)alloc((size_t)Bn * Cc * 4);
    float* ca1v   = (float*)alloc((size_t)Bn * Cc * 4);
    float* ca2v   = (float*)alloc((size_t)Bn * Cc * 4);

    // ---- channel attention path
    gap_kernel<<<Bn * Cc, 256, 0, stream>>>(x, gapb);
    ca_kernel<<<1, 192, 0, stream>>>(gapb, ca1w, ca1b, ca2w, ca2b, ca1v, ca2v);

    // ---- layer norms
    ln_kernel<<<Np / 256, 256, 0, stream>>>(x, n1w, n1b, n2w, n2b, xn, xm);

    // ---- KBA branch
    conv1x1_kernel<<<dim3(Np / 256, HIDc / 16), 256, 0, stream>>>(xn, kdw1, nullptr, x1p, Cc, HIDc);
    dw3x3_kernel<<<(Bn * HIDc * HWp) / 256, 256, 0, stream>>>(x1p, kdw2, x1, HIDc);
    conv1x1_kernel<<<dim3(Np / 256, HIDc / 16), 256, 0, stream>>>(xn, kc1a, nullptr, ufp, Cc, HIDc);
    dw3x3_kernel<<<(Bn * HIDc * HWp) / 256, 256, 0, stream>>>(ufp, kc1b, ufb, HIDc);
    gconv3x3_kernel<<<(Bn * INTERCc * HWp) / 256, 256, 0, stream>>>(xn, c2aw, c2ab, apre);
    att_kernel<<<Np / 256, 256, 0, stream>>>(apre, xn, c2bw, c2bb, c211w, c211b, attg, atth);
    cvt_kernel<<<(NSETc * 3456 + 255) / 256, 256, 0, stream>>>(kw, kwh, NSETc * 3456);
    cvt_kernel<<<(NSETc * HIDc + 255) / 256, 256, 0, stream>>>(kb, kbh, NSETc * HIDc);
    kba_kernel<<<(Np / 16) / KBA_WAVES, 32 * KBA_WAVES, 0, stream>>>(atth, kwh, kbh, ufb, x1, ga1, x2g);
    conv1x1_kernel<<<dim3(Np / 256, Cc / 16), 256, 0, stream>>>(x2g, kproj, nullptr, kbaout, HIDc, Cc);

    // ---- MDTA branch
    conv1x1_kernel<<<dim3(Np / 256, 144 / 16), 256, 0, stream>>>(xm, qkvw, nullptr, qkvp, Cc, 144);
    dw3x3_kernel<<<(Bn * 144 * HWp) / 256, 256, 0, stream>>>(qkvp, qkvdw, qkv, 144);
    mdta_kernel<<<Bn * 8, 256, 0, stream>>>(qkv, temp, attno);
    conv1x1_kernel<<<dim3(Np / 256, Cc / 16), 256, 0, stream>>>(attno, mprojw, nullptr, mdtao, Cc, Cc);

    // ---- combine
    combine_kernel<<<(Bn * Cc * HWp) / 256, 256, 0, stream>>>(x, kbaout, mdtao, ca1v, ca2v, (float*)d_out);
}
